// DeformableTransformer_19344532701351
// MI455X (gfx1250) — matrix operand order repaired and checked by reading, not compiled, not run on hardware
//
#include <hip/hip_runtime.h>
#include <hip/hip_bf16.h>

// ---------------- problem constants (match reference) ----------------
#define NUM_LAYERS 2
#define DMODEL 256
#define NHEAD  8
#define NPOINT 4
#define NLVL   4
#define DHEAD  32          // DMODEL / NHEAD
#define DFF    1024
#define BATCH  2
#define S_TOT  13294       // 100*100 + 50*50 + 25*25 + 13*13
#define BSROWS (BATCH * S_TOT)

typedef __bf16 bf16_t;
typedef __attribute__((ext_vector_type(16))) __bf16        v16bf;
typedef __attribute__((ext_vector_type(8)))  float         v8f;
typedef __attribute__((ext_vector_type(4)))  unsigned int  u32x4;
typedef __attribute__((ext_vector_type(4)))  float         f32x4;

union FragBF { v16bf v; u32x4 q[2]; };

// ---------------- bf16 WMMA GEMM: C[M,N] = A[M,K] * W[K,N] + bias -----------
// block: 256 threads = 8 waves; macro tile 128x64; wave tile 32x32; BK = 32.
// Double-buffered LDS (one barrier per K-step), branch-free clamped loads,
// 4x v_wmma_f32_16x16x32_bf16 per wave per K-step.
#define BM 128
#define BN 64
#define BK 32
#define LDT 40   // padded LDS row stride (bf16 elems); 80B rows keep 16B alignment

__global__ __launch_bounds__(256)
void gemm_wmma_bf16(const float* __restrict__ A, const float* __restrict__ W,
                    const float* __restrict__ bias, float* __restrict__ C,
                    int M, int N, int K, int relu)
{
    __shared__ __align__(16) bf16_t As[2][BM * LDT];  // 2 x 10240 B, [m][k]
    __shared__ __align__(16) bf16_t Bt[2][BN * LDT];  // 2 x  5120 B, [n][k] (transposed)

    const int tid  = threadIdx.x;
    const int lane = tid & 31;
    const int wave = tid >> 5;
    const int wm   = wave >> 1;          // 0..3 : 32-row strip
    const int wn   = wave & 1;           // 0..1 : 32-col strip
    const int r    = lane & 15;
    const int hi   = lane >> 4;

    const int mBase = blockIdx.x * BM;
    const int nBase = blockIdx.y * BN;

    // A tile loader: 128x32 fp32, 16 floats/thread (contiguous 16)
    const int aRow  = tid >> 1;                    // 0..127
    const int aCol  = (tid & 1) * 16;              // 0 or 16
    const int aRowG = min(mBase + aRow, M - 1);    // clamp: branch-free, always in-bounds
    // B tile loader: 32x64 fp32, 8 floats/thread, written transposed
    const int bRow  = tid >> 3;                    // 0..31 (k)
    const int bCol  = (tid & 7) * 8;               // 0..56 (n)

    v8f acc[2][2] = {{{}, {}}, {{}, {}}};

    f32x4 ra[4];
    f32x4 rb[2];

    auto gload = [&](int k0) {
        const float* pa = A + (long)aRowG * K + k0 + aCol;
        ra[0] = *(const f32x4*)(pa);
        ra[1] = *(const f32x4*)(pa + 4);
        ra[2] = *(const f32x4*)(pa + 8);
        ra[3] = *(const f32x4*)(pa + 12);
        __builtin_prefetch(pa + BK, 0, 1);          // global_prefetch_b8 (speculative)
        const float* pw = W + (long)(k0 + bRow) * N + nBase + bCol;
        rb[0] = *(const f32x4*)(pw);
        rb[1] = *(const f32x4*)(pw + 4);
    };

    auto sstore = [&](int buf) {
        bf16_t* da = &As[buf][aRow * LDT + aCol];
        da[0]  = (bf16_t)ra[0].x; da[1]  = (bf16_t)ra[0].y;
        da[2]  = (bf16_t)ra[0].z; da[3]  = (bf16_t)ra[0].w;
        da[4]  = (bf16_t)ra[1].x; da[5]  = (bf16_t)ra[1].y;
        da[6]  = (bf16_t)ra[1].z; da[7]  = (bf16_t)ra[1].w;
        da[8]  = (bf16_t)ra[2].x; da[9]  = (bf16_t)ra[2].y;
        da[10] = (bf16_t)ra[2].z; da[11] = (bf16_t)ra[2].w;
        da[12] = (bf16_t)ra[3].x; da[13] = (bf16_t)ra[3].y;
        da[14] = (bf16_t)ra[3].z; da[15] = (bf16_t)ra[3].w;
        bf16_t* db = &Bt[buf][0];
        db[(bCol + 0) * LDT + bRow] = (bf16_t)rb[0].x;
        db[(bCol + 1) * LDT + bRow] = (bf16_t)rb[0].y;
        db[(bCol + 2) * LDT + bRow] = (bf16_t)rb[0].z;
        db[(bCol + 3) * LDT + bRow] = (bf16_t)rb[0].w;
        db[(bCol + 4) * LDT + bRow] = (bf16_t)rb[1].x;
        db[(bCol + 5) * LDT + bRow] = (bf16_t)rb[1].y;
        db[(bCol + 6) * LDT + bRow] = (bf16_t)rb[1].z;
        db[(bCol + 7) * LDT + bRow] = (bf16_t)rb[1].w;
    };

    auto compute = [&](int buf) {
        // A fragments: rows wm*32 + {0,16} + r; elems {kb..kb+7, kb+16..kb+23}, kb = hi*8
        FragBF a0, a1, b0, b1;
        const bf16_t* pA0 = &As[buf][(wm * 32 + r) * LDT + hi * 8];
        const bf16_t* pA1 = pA0 + 16 * LDT;
        a0.q[0] = *(const u32x4*)(pA0);
        a0.q[1] = *(const u32x4*)(pA0 + 16);
        a1.q[0] = *(const u32x4*)(pA1);
        a1.q[1] = *(const u32x4*)(pA1 + 16);
        // B fragments: cols wn*32 + {0,16} + r; elems {kb..kb+7, kb+16..kb+23}, kb = hi*8
        const bf16_t* pB0 = &Bt[buf][(wn * 32 + r) * LDT + hi * 8];
        const bf16_t* pB1 = pB0 + 16 * LDT;
        b0.q[0] = *(const u32x4*)(pB0);
        b0.q[1] = *(const u32x4*)(pB0 + 16);
        b1.q[0] = *(const u32x4*)(pB1);
        b1.q[1] = *(const u32x4*)(pB1 + 16);

        acc[0][0] = __builtin_amdgcn_wmma_f32_16x16x32_bf16(false, a0.v, false, b0.v,
                                                            (short)0, acc[0][0], false, false);
        acc[0][1] = __builtin_amdgcn_wmma_f32_16x16x32_bf16(false, a0.v, false, b1.v,
                                                            (short)0, acc[0][1], false, false);
        acc[1][0] = __builtin_amdgcn_wmma_f32_16x16x32_bf16(false, a1.v, false, b0.v,
                                                            (short)0, acc[1][0], false, false);
        acc[1][1] = __builtin_amdgcn_wmma_f32_16x16x32_bf16(false, a1.v, false, b1.v,
                                                            (short)0, acc[1][1], false, false);
    };

    // software pipeline: preload tile 0, then one barrier per K-step
    gload(0);
    sstore(0);
    __syncthreads();
    int cur = 0;
    for (int k0 = 0; k0 < K; k0 += BK) {
        const bool hasNext = (k0 + BK) < K;
        if (hasNext) gload(k0 + BK);      // global loads overlap WMMA below
        compute(cur);
        if (hasNext) {
            sstore(cur ^ 1);              // safe: last read of buf cur^1 was fenced
            __syncthreads();              // by this barrier one iteration ago
            cur ^= 1;
        }
    }

    // C layout: lane r = column, hi selects M 0..7 vs 8..15, vgpr = row-in-8
    const int col0 = nBase + wn * 32 + r;
    const float bia0 = bias[col0];
    const float bia1 = bias[col0 + 16];
    for (int sub = 0; sub < 2; ++sub) {
        for (int rv = 0; rv < 8; ++rv) {
            const int row = mBase + wm * 32 + sub * 16 + hi * 8 + rv;
            if (row < M) {
                float o0 = acc[sub][0][rv] + bia0;
                float o1 = acc[sub][1][rv] + bia1;
                if (relu) { o0 = fmaxf(o0, 0.f); o1 = fmaxf(o1, 0.f); }
                C[(long)row * N + col0]      = o0;
                C[(long)row * N + col0 + 16] = o1;
            }
        }
    }
}

// ---------------- deformable attention sampling (softmax fused) -------------
__constant__ int c_lvlStart[NLVL] = {0, 10000, 12500, 13125};
__constant__ int c_lvlH[NLVL]     = {100, 50, 25, 13};
__constant__ int c_lvlW[NLVL]     = {100, 50, 25, 13};

__global__ __launch_bounds__(256)
void msda_sample(const float* __restrict__ value,  // [B,S,NH,DH]
                 const float* __restrict__ offr,   // [B*S, NH*NL*NP*2]
                 const float* __restrict__ attnr,  // [B*S, NH*NL*NP]
                 const float* __restrict__ vr,     // [B, NL, 2] (x,y)
                 float* __restrict__ out)          // [B,S,D], D idx = h*DH+dh
{
    const int lane = threadIdx.x & 31;                       // dh channel
    const int wid  = blockIdx.x * (blockDim.x >> 5) + (threadIdx.x >> 5);
    const int h    = wid & (NHEAD - 1);
    const int rest = wid >> 3;
    const int q    = rest % S_TOT;
    const int b    = rest / S_TOT;
    if (b >= BATCH) return;

    // reference point of this query (its own level grid position)
    const int lvlq = (q >= 13125) ? 3 : (q >= 12500) ? 2 : (q >= 10000) ? 1 : 0;
    const int qo = q - c_lvlStart[lvlq];
    const int Wq = c_lvlW[lvlq];
    const int iy = qo / Wq, ix = qo % Wq;
    const float refx = (ix + 0.5f) / (vr[(b * NLVL + lvlq) * 2 + 0] * (float)Wq);
    const float refy = (iy + 0.5f) / (vr[(b * NLVL + lvlq) * 2 + 1] * (float)c_lvlH[lvlq]);

    const long rowq = (long)b * S_TOT + q;

    // softmax over the 16 (level,point) logits of this head
    const float* ap = attnr + rowq * (NHEAD * NLVL * NPOINT) + h * (NLVL * NPOINT);
    float w16[16];
    float mx = -1e30f;
    for (int i = 0; i < 16; ++i) { w16[i] = ap[i]; mx = fmaxf(mx, w16[i]); }
    float ssum = 0.f;
    for (int i = 0; i < 16; ++i) { w16[i] = __expf(w16[i] - mx); ssum += w16[i]; }
    const float inv = 1.f / ssum;

    const float* op = offr + rowq * (NHEAD * NLVL * NPOINT * 2) + h * (NLVL * NPOINT * 2);
    float acc = 0.f;
    for (int l = 0; l < NLVL; ++l) {
        const int H_ = c_lvlH[l], W_ = c_lvlW[l], st = c_lvlStart[l];
        const float rx = refx * vr[(b * NLVL + l) * 2 + 0];
        const float ry = refy * vr[(b * NLVL + l) * 2 + 1];
        const float* vbase = value + (((long)b * S_TOT + st) * NHEAD + h) * DHEAD + lane;
        for (int p = 0; p < NPOINT; ++p) {
            const float ox = op[l * NPOINT * 2 + p * 2 + 0];
            const float oy = op[l * NPOINT * 2 + p * 2 + 1];
            const float x = (rx + ox / (float)W_) * (float)W_ - 0.5f;
            const float y = (ry + oy / (float)H_) * (float)H_ - 0.5f;
            const float x0f = floorf(x), y0f = floorf(y);
            const int   x0 = (int)x0f, y0 = (int)y0f;
            const float fx = x - x0f, fy = y - y0f;
            float s = 0.f;
            auto corner = [&](int xi, int yi, float w) {
                if (xi >= 0 && xi < W_ && yi >= 0 && yi < H_) {
                    // stride NH*DH floats per spatial index -> 128B coalesced per wave
                    s += w * vbase[(long)(yi * W_ + xi) * (NHEAD * DHEAD)];
                }
            };
            corner(x0,     y0,     (1.f - fx) * (1.f - fy));
            corner(x0 + 1, y0,     fx         * (1.f - fy));
            corner(x0,     y0 + 1, (1.f - fx) * fy);
            corner(x0 + 1, y0 + 1, fx         * fy);
            acc += (w16[l * NPOINT + p] * inv) * s;
        }
    }
    out[rowq * DMODEL + h * DHEAD + lane] = acc;
}

// ---------------- fused residual + LayerNorm (wave per row, wave32) ---------
__global__ __launch_bounds__(256)
void add_layernorm(const float* __restrict__ x, const float* __restrict__ yv,
                   const float* __restrict__ g, const float* __restrict__ bta,
                   float* __restrict__ out, int rows)
{
    const int lane = threadIdx.x & 31;
    const int row  = blockIdx.x * (blockDim.x >> 5) + (threadIdx.x >> 5);
    if (row >= rows) return;
    const long base = (long)row * DMODEL + lane * 8;
    f32x4 a0 = *(const f32x4*)(x  + base);
    f32x4 a1 = *(const f32x4*)(x  + base + 4);
    f32x4 b0 = *(const f32x4*)(yv + base);
    f32x4 b1 = *(const f32x4*)(yv + base + 4);
    f32x4 v0 = a0 + b0, v1 = a1 + b1;
    float s  = v0.x + v0.y + v0.z + v0.w + v1.x + v1.y + v1.z + v1.w;
    float s2 = v0.x*v0.x + v0.y*v0.y + v0.z*v0.z + v0.w*v0.w
             + v1.x*v1.x + v1.y*v1.y + v1.z*v1.z + v1.w*v1.w;
    for (int m = 16; m; m >>= 1) {
        s  += __shfl_xor(s,  m, 32);
        s2 += __shfl_xor(s2, m, 32);
    }
    const float mean = s * (1.f / DMODEL);
    const float var  = s2 * (1.f / DMODEL) - mean * mean;
    const float rstd = rsqrtf(var + 1e-5f);
    const int c = lane * 8;
    f32x4 g0 = *(const f32x4*)(g + c),   g1 = *(const f32x4*)(g + c + 4);
    f32x4 t0 = *(const f32x4*)(bta + c), t1 = *(const f32x4*)(bta + c + 4);
    f32x4 o0 = (v0 - mean) * rstd * g0 + t0;
    f32x4 o1 = (v1 - mean) * rstd * g1 + t1;
    *(f32x4*)(out + base)     = o0;
    *(f32x4*)(out + base + 4) = o1;
}

// ---------------- orchestration --------------------------------------------
extern "C" void kernel_launch(void* const* d_in, const int* in_sizes, int n_in,
                              void* d_out, int out_size, void* d_ws, size_t ws_size,
                              hipStream_t stream)
{
    const float* src    = (const float*)d_in[0];
    // d_in[1] = spatial_shapes (int64) — compile-time known, unused
    const float* vr     = (const float*)d_in[2];
    const float* W_off  = (const float*)d_in[3];
    const float* b_off  = (const float*)d_in[4];
    const float* W_attn = (const float*)d_in[5];
    const float* b_attn = (const float*)d_in[6];
    const float* W_val  = (const float*)d_in[7];
    const float* b_val  = (const float*)d_in[8];
    const float* W_out  = (const float*)d_in[9];
    const float* b_out  = (const float*)d_in[10];
    const float* ln1g   = (const float*)d_in[11];
    const float* ln1b   = (const float*)d_in[12];
    const float* W1     = (const float*)d_in[13];
    const float* b1     = (const float*)d_in[14];
    const float* W2     = (const float*)d_in[15];
    const float* b2     = (const float*)d_in[16];
    const float* ln2g   = (const float*)d_in[17];
    const float* ln2b   = (const float*)d_in[18];

    float* ws = (float*)d_ws;
    const long BSD = (long)BSROWS * DMODEL;
    float* val  = ws;
    float* off  = val  + BSD;
    float* attn = off  + BSD;
    float* samp = attn + (long)BSROWS * 128;
    float* tmp  = samp + BSD;
    float* h1   = tmp  + BSD;
    float* xA   = h1   + (long)BSROWS * DFF;
    float* xB   = xA   + BSD;

    const dim3 blk(256);
    const int  mT = (BSROWS + BM - 1) / BM;
    auto gemm = [&](const float* A, const float* Wm, const float* bi, float* Cm,
                    int N, int K, int relu) {
        hipLaunchKernelGGL(gemm_wmma_bf16, dim3(mT, N / BN), blk, 0, stream,
                           A, Wm, bi, Cm, BSROWS, N, K, relu);
    };

    const float* cur = src;
    for (int i = 0; i < NUM_LAYERS; ++i) {
        gemm(cur, W_val  + (long)i * DMODEL * DMODEL, b_val  + i * DMODEL, val, DMODEL, DMODEL, 0);
        gemm(cur, W_off  + (long)i * DMODEL * 256,    b_off  + i * 256,    off, 256,    DMODEL, 0);
        gemm(cur, W_attn + (long)i * DMODEL * 128,    b_attn + i * 128,    attn, 128,   DMODEL, 0);

        hipLaunchKernelGGL(msda_sample, dim3(BSROWS * NHEAD / 8), blk, 0, stream,
                           val, off, attn, vr, samp);

        gemm(samp, W_out + (long)i * DMODEL * DMODEL, b_out + i * DMODEL, tmp, DMODEL, DMODEL, 0);
        hipLaunchKernelGGL(add_layernorm, dim3((BSROWS + 7) / 8), blk, 0, stream,
                           cur, tmp, ln1g + i * DMODEL, ln1b + i * DMODEL, xA, BSROWS);

        gemm(xA, W1 + (long)i * DMODEL * DFF, b1 + i * DFF, h1, DFF, DMODEL, 1);
        gemm(h1, W2 + (long)i * DFF * DMODEL, b2 + i * DMODEL, tmp, DMODEL, DFF, 0);

        float* dst = (i == NUM_LAYERS - 1) ? (float*)d_out : xB;
        hipLaunchKernelGGL(add_layernorm, dim3((BSROWS + 7) / 8), blk, 0, stream,
                           xA, tmp, ln2g + i * DMODEL, ln2b + i * DMODEL, dst, BSROWS);
        cur = xB;
    }
}
